// SRLModel_attn_57475252355700
// MI455X (gfx1250) — matrix operand order repaired and checked by reading, not compiled, not run on hardware
//
#include <hip/hip_runtime.h>
#include <stdint.h>

// Problem dims (B,S,P,D,L) = (8,512,64,256,64)
#define Bn 8
#define Sn 512
#define Pn 64
#define Dn 256
#define Ln 64
#define BPn 512      // B*P
#define BSn 4096     // B*S
#define N1n 16384    // L*D
#define LDP 264      // padded LDS row stride in bf16 elements (528B = 132 dwords)

typedef __attribute__((ext_vector_type(16))) __bf16 bf16x16;
typedef __attribute__((ext_vector_type(8)))  float  f32x8;
typedef __attribute__((ext_vector_type(4)))  unsigned int u32x4;

__device__ __forceinline__ unsigned short f2bf(float f) {
  union { float f; unsigned int u; } c; c.f = f;
  unsigned int u = c.u;
  u += 0x7FFFu + ((u >> 16) & 1u);          // round-to-nearest-even
  return (unsigned short)(u >> 16);
}

union FragU { bf16x16 v; u32x4 q[2]; };

// ---- async global -> LDS staging (ASYNCcnt path) ----
__device__ __forceinline__ void async_ld_b128(unsigned lds_byte_addr, const void* gptr) {
  asm volatile("global_load_async_to_lds_b128 %0, %1, off"
               :: "v"(lds_byte_addr), "v"(gptr)
               : "memory");
}
__device__ __forceinline__ void wait_async0() {
  asm volatile("s_wait_asynccnt 0x0" ::: "memory");
}

// A fragment (16x32, row-major MxK): lane l: m=l&15, h=l>>4;
// elems 0..7 -> K=k0+h*8+i ; elems 8..15 -> K=k0+16+h*8+i  (two 16B chunks)
__device__ __forceinline__ bf16x16 frag_a(const unsigned short* __restrict__ base,
                                          int ld, int row0, int k0, int lane) {
  const int m = lane & 15;
  const int h = lane >> 4;
  const unsigned short* p = base + (size_t)(row0 + m) * ld + k0 + h * 8;
  FragU f;
  f.q[0] = *(const u32x4*)(p);
  f.q[1] = *(const u32x4*)(p + 16);
  return f.v;
}

// B fragment (32x16 KxN) from B^T stored row-major (N x K): lane l: n=l&15, h=l>>4;
// elems 0..7 -> K=k0+h*8+i ; elems 8..15 -> K=k0+16+h*8+i  (two 16B chunks)
__device__ __forceinline__ bf16x16 frag_b(const unsigned short* __restrict__ bt,
                                          int ld, int n0, int k0, int lane) {
  const int n = lane & 15;
  const int h = lane >> 4;
  const unsigned short* p = bt + (size_t)(n0 + n) * ld + k0 + h * 8;
  FragU f;
  f.q[0] = *(const u32x4*)(p);
  f.q[1] = *(const u32x4*)(p + 16);
  return f.v;
}

#define WMMA_BF16(ACC, A, Bf) \
  ACC = __builtin_amdgcn_wmma_f32_16x16x32_bf16(false, (A), false, (Bf), (short)0, ACC, false, false)

// ---------------- prep kernels ----------------

__global__ void prep_span_bf16(const float* __restrict__ span,
                               unsigned short* __restrict__ spanB) {
  int i = blockIdx.x * 256 + threadIdx.x;          // BS*D = 1,048,576
  spanB[i] = f2bf(span[i]);
}

__global__ void prep_pred(const float* __restrict__ span,
                          const int* __restrict__ preds,
                          float* __restrict__ predF,
                          unsigned short* __restrict__ predB) {
  int bp = blockIdx.x;                              // 512
  int d  = threadIdx.x;                             // 256
  int b  = bp >> 6;
  int idx = preds[bp];
  float v = span[((size_t)b * Sn + idx) * Dn + d];
  predF[bp * Dn + d] = v;
  predB[bp * Dn + d] = f2bf(v);
}

__global__ void prep_w1t(const float* __restrict__ W1,
                         unsigned short* __restrict__ w1t) {
  int n = blockIdx.x;                               // 16384 = l*256+e
  int d = threadIdx.x;                              // 256
  int l = n >> 8, e = n & 255;
  w1t[(size_t)n * Dn + d] = f2bf(W1[(size_t)l * Dn * Dn + (size_t)d * Dn + e]);
}

__global__ void prep_w2lo(const float* __restrict__ W2,
                          float* __restrict__ w2loT) {
  int n = blockIdx.x * 256 + threadIdx.x;           // 16384
  int l = n >> 8, e = n & 255;
  w2loT[n] = W2[(size_t)(Dn + e) * Ln + l];
}

__global__ void second_pred_kernel(const float* __restrict__ predF,
                                   const float* __restrict__ W2,
                                   float* __restrict__ second_pred) {
  __shared__ float pr[Dn];
  int bp = blockIdx.x;                              // 512
  int l  = threadIdx.x;                             // 64
  for (int j = l; j < Dn; j += 64) pr[j] = predF[bp * Dn + j];
  __syncthreads();
  float acc = 0.f;
  for (int d = 0; d < Dn; ++d) acc = fmaf(pr[d], W2[d * Ln + l], acc);
  second_pred[bp * Ln + l] = acc;
}

// relu(x @ W + bias) . w  for one row per block (256 threads)
__global__ void ffnn_score_kernel(const float* __restrict__ X, // rows x D
                                  const float* __restrict__ W, // D x D
                                  const float* __restrict__ bias,
                                  const float* __restrict__ w,
                                  float* __restrict__ out) {
  __shared__ float xr[Dn];
  __shared__ float red[256];
  int row = blockIdx.x;
  int t = threadIdx.x;
  xr[t] = X[(size_t)row * Dn + t];
  __syncthreads();
  float h = bias[t];
  for (int d = 0; d < Dn; ++d) h = fmaf(xr[d], W[d * Dn + t], h);
  red[t] = fmaxf(h, 0.f) * w[t];
  __syncthreads();
  for (int s = 128; s > 0; s >>= 1) {
    if (t < s) red[t] += red[t + s];
    __syncthreads();
  }
  if (t == 0) out[row] = red[0];
}

__global__ void labels_out_kernel(const int* __restrict__ labels,
                                  float* __restrict__ out_tail) {
  int i = blockIdx.x * 256 + threadIdx.x;           // 262144
  out_tail[i] = (float)labels[i];
}

// ---------------- GEMM 1: tmp2[bp, l*256+e] = pred @ W1t^T + W2lo ----------------
// M=512(bp), N=16384(l,e), K=256(d). Block = 4 waves; block tile 32(M) x 256(N);
// wave strip 32x64 (2 m-tiles x 4 n-tiles = 8 accumulators).
// Shared A panel (pred rows m0..m0+31) async-staged to LDS, reused by all 4 waves.
__global__ __launch_bounds__(128) void srl_gemm1(
    const unsigned short* __restrict__ predB,   // [512,256] bf16
    const unsigned short* __restrict__ w1t,     // [16384,256] bf16 (N x K)
    const float* __restrict__ w2loT,            // [16384]
    unsigned short* __restrict__ tmp2) {        // [512,16384] bf16
  __shared__ alignas(16) unsigned short shA[32 * LDP];   // 16.9 KB
  const int tid  = threadIdx.x;
  const int lane = tid & 31;
  const int wave = tid >> 5;
  const int m0 = blockIdx.x * 32;
  const int n0 = blockIdx.y * 256 + wave * 64;

  // async-stage A panel: 32 rows x 512B = 1024 x 16B chunks, 8 per thread
  {
    const unsigned base = (unsigned)(uintptr_t)(&shA[0]);
#pragma unroll
    for (int i = 0; i < 8; ++i) {
      int c = tid + i * 128;
      int row = c >> 5, col = c & 31;
      async_ld_b128(base + row * (LDP * 2) + col * 16,
                    predB + (size_t)(m0 + row) * Dn + col * 8);
    }
  }
  __builtin_prefetch(w1t + (size_t)n0 * Dn, 0, 1);
  wait_async0();
  __syncthreads();

  f32x8 acc[2][4] = {};
#pragma unroll
  for (int k0 = 0; k0 < Dn; k0 += 32) {
    bf16x16 a0 = frag_a(shA, LDP, 0, k0, lane);     // ds_load_b128 x2
    bf16x16 a1 = frag_a(shA, LDP, 16, k0, lane);
#pragma unroll
    for (int t = 0; t < 4; ++t) {
      bf16x16 bf = frag_b(w1t, Dn, n0 + t * 16, k0, lane);
      WMMA_BF16(acc[0][t], a0, bf);
      WMMA_BF16(acc[1][t], a1, bf);
    }
  }

  const int half = lane >> 4;
  const int nn = lane & 15;
#pragma unroll
  for (int mt = 0; mt < 2; ++mt) {
#pragma unroll
    for (int t = 0; t < 4; ++t) {
      union { f32x8 v; float f[8]; } u;
      u.v = acc[mt][t];
      const int n = n0 + t * 16 + nn;
      const float w2 = w2loT[n];
#pragma unroll
      for (int r = 0; r < 8; ++r) {
        const int m = m0 + mt * 16 + half * 8 + r;
        tmp2[(size_t)m * N1n + n] = f2bf(u.f[r] + w2);
      }
    }
  }
}

// ---------------- GEMM 2: scores[bp, s, l] with fused epilogue ----------------
// Per bp: C = spanB[b] (512x256) @ tmp2[bp]^T (256x64). Block = (bp, sblk of 128 rows),
// 4 waves x 32-row strips. Shared B panel tmp2[bp] (64x256) async-staged to LDS.
__global__ __launch_bounds__(128) void srl_gemm2(
    const unsigned short* __restrict__ spanB,       // [4096,256] bf16
    const unsigned short* __restrict__ tmp2,        // [512][64][256] bf16 (N x K per bp)
    const float* __restrict__ second_pred,          // [512,64]
    const float* __restrict__ bvec,                 // [64]
    const float* __restrict__ p_score,              // [512]
    const float* __restrict__ a_score,              // [4096]
    float* __restrict__ out) {                      // [512*512*64]
  __shared__ alignas(16) unsigned short shB[Ln * LDP];   // 33.8 KB
  const int tid  = threadIdx.x;
  const int lane = tid & 31;
  const int wave = tid >> 5;
  const int bp   = blockIdx.x;                      // 512
  const int sblk = blockIdx.y;                      // 4
  const int b    = bp >> 6;
  const int s0   = sblk * 128 + wave * 32;

  const unsigned short* A  = spanB + (size_t)b * Sn * Dn;
  const unsigned short* Bt = tmp2  + (size_t)bp * N1n;

  // async-stage B panel: 64 rows x 512B = 2048 x 16B chunks, 16 per thread
  {
    const unsigned base = (unsigned)(uintptr_t)(&shB[0]);
#pragma unroll
    for (int i = 0; i < 16; ++i) {
      int c = tid + i * 128;
      int row = c >> 5, col = c & 31;
      async_ld_b128(base + row * (LDP * 2) + col * 16,
                    Bt + (size_t)row * Dn + col * 8);
    }
  }
  __builtin_prefetch(A + (size_t)s0 * Dn, 0, 1);
  wait_async0();
  __syncthreads();

  f32x8 acc[2][4] = {};
#pragma unroll
  for (int k0 = 0; k0 < Dn; k0 += 32) {
    bf16x16 a0 = frag_a(A, Dn, s0, k0, lane);       // global_load_b128 x2
    bf16x16 a1 = frag_a(A, Dn, s0 + 16, k0, lane);
#pragma unroll
    for (int t = 0; t < 4; ++t) {
      bf16x16 bf = frag_b(shB, LDP, t * 16, k0, lane);   // ds_load_b128 x2
      WMMA_BF16(acc[0][t], a0, bf);
      WMMA_BF16(acc[1][t], a1, bf);
    }
  }

  const int half = lane >> 4;
  const int nn = lane & 15;
  const float ps = p_score[bp];
  float* outbp = out + (size_t)bp * Sn * Ln;
#pragma unroll
  for (int mt = 0; mt < 2; ++mt) {
#pragma unroll
    for (int t = 0; t < 4; ++t) {
      union { f32x8 v; float f[8]; } u;
      u.v = acc[mt][t];
      const int l = t * 16 + nn;
      const float col = second_pred[bp * Ln + l] + bvec[l] + ps;
#pragma unroll
      for (int r = 0; r < 8; ++r) {
        const int s = s0 + mt * 16 + half * 8 + r;
        float v = u.f[r] + col + a_score[b * Sn + s];
        if (l == Ln - 1) v = 0.f;                   // combine_score[..., -1] = 0
        outbp[(size_t)s * Ln + l] = v;
      }
    }
  }
}

// ---------------- host launch ----------------

extern "C" void kernel_launch(void* const* d_in, const int* in_sizes, int n_in,
                              void* d_out, int out_size, void* d_ws, size_t ws_size,
                              hipStream_t stream) {
  const float* span  = (const float*)d_in[0];
  const float* Wp_ff = (const float*)d_in[1];
  const float* bp_ff = (const float*)d_in[2];
  const float* Wa_ff = (const float*)d_in[3];
  const float* ba_ff = (const float*)d_in[4];
  const float* w_p   = (const float*)d_in[5];
  const float* w_a   = (const float*)d_in[6];
  const float* W1    = (const float*)d_in[7];
  const float* W2    = (const float*)d_in[8];
  const float* bvec  = (const float*)d_in[9];
  const int* preds   = (const int*)d_in[10];
  const int* labels  = (const int*)d_in[11];
  float* out = (float*)d_out;

  // workspace layout (bytes)
  char* ws = (char*)d_ws;
  float*          predF  = (float*)(ws + 0);                   // 512*256*4  = 524288
  unsigned short* predB  = (unsigned short*)(ws + 524288);     // 512*256*2  = 262144
  unsigned short* spanB  = (unsigned short*)(ws + 786432);     // 4096*256*2 = 2097152
  unsigned short* w1t    = (unsigned short*)(ws + 2883584);    // 16384*256*2= 8388608
  float*          w2loT  = (float*)(ws + 11272192);            // 16384*4    = 65536
  float*          spred  = (float*)(ws + 11337728);            // 512*64*4   = 131072
  float*          pscore = (float*)(ws + 11468800);            // 512*4
  float*          ascore = (float*)(ws + 11470848);            // 4096*4
  unsigned short* tmp2   = (unsigned short*)(ws + 11487232);   // 512*16384*2 = 16777216
  (void)in_sizes; (void)n_in; (void)out_size; (void)ws_size;

  prep_span_bf16<<<BSn, 256, 0, stream>>>(span, spanB);
  prep_pred<<<BPn, 256, 0, stream>>>(span, preds, predF, predB);
  prep_w1t<<<N1n, 256, 0, stream>>>(W1, w1t);
  prep_w2lo<<<N1n / 256, 256, 0, stream>>>(W2, w2loT);
  second_pred_kernel<<<BPn, 64, 0, stream>>>(predF, W2, spred);
  ffnn_score_kernel<<<BPn, 256, 0, stream>>>(predF, Wp_ff, bp_ff, w_p, pscore);
  ffnn_score_kernel<<<BSn, 256, 0, stream>>>(span, Wa_ff, ba_ff, w_a, ascore);

  srl_gemm1<<<dim3(BPn / 32, N1n / 256), 128, 0, stream>>>(predB, w1t, w2loT, tmp2);
  srl_gemm2<<<dim3(BPn, Sn / 128), 128, 0, stream>>>(spanB, tmp2, spred, bvec, pscore, ascore, out);

  labels_out_kernel<<<(Bn * Pn * Sn) / 256, 256, 0, stream>>>(labels, out + (size_t)BPn * Sn * Ln);
}
